// DotProductAttetion_781684048501
// MI455X (gfx1250) — matrix operand order repaired and checked by reading, not compiled, not run on hardware
//
#include <hip/hip_runtime.h>
#include <stdint.h>

// ---- CDNA5 WMMA types -------------------------------------------------------
typedef __attribute__((ext_vector_type(16))) __bf16        bf16x16;
typedef __attribute__((ext_vector_type(8)))  float         f32x8;
typedef __attribute__((ext_vector_type(8)))  unsigned int  u32x8;

#define S_LEN   2048
#define NBATCH  16
#define DHEAD   128
#define BM      64          // rows per workgroup (4 waves x 16 rows)
#define NTHREADS 128

#define KS_STRIDE 134       // ushorts per row (67 dwords) -> conflict-free A reads
#define QS_STRIDE 134       // ushorts per row (67 dwords) -> conflict-free B reads
#define VT_STRIDE 70        // ushorts per row (35 dwords) -> conflict-free B reads
#define PS_STRIDE 38        // ushorts per row (19 dwords) -> conflict-free A reads

__device__ __forceinline__ unsigned short f2bf(float x) {
    union { float f; unsigned int u; } c; c.f = x;
    unsigned int r = c.u + 0x7fffu + ((c.u >> 16) & 1u);   // round-to-nearest-even
    return (unsigned short)(r >> 16);
}
__device__ __forceinline__ unsigned int pack2bf(float lo, float hi) {
    return ((unsigned int)f2bf(hi) << 16) | (unsigned int)f2bf(lo);
}

// A fragment (16x32 bf16): lane m holds row m; dword pattern (v<4? v : v+4) + 4*half
__device__ __forceinline__ bf16x16 ldA(const unsigned short* row, int kc, int h) {
    const unsigned int* p = reinterpret_cast<const unsigned int*>(row) + kc * 16;
    u32x8 t;
#pragma unroll
    for (int v = 0; v < 8; ++v) t[v] = p[(v < 4 ? v : v + 4) + 4 * h];
    return __builtin_bit_cast(bf16x16, t);
}

// B fragment (32x16 bf16): lane n holds col n; dword pattern (v<4? v : v+4) + 4*half
__device__ __forceinline__ bf16x16 ldB(const unsigned short* row, int kc, int h) {
    const unsigned int* p = reinterpret_cast<const unsigned int*>(row) + kc * 16;
    u32x8 t;
#pragma unroll
    for (int v = 0; v < 8; ++v) t[v] = p[(v < 4 ? v : v + 4) + 4 * h];
    return __builtin_bit_cast(bf16x16, t);
}

__device__ __forceinline__ bf16x16 ldB_at(const unsigned short* row, int dwbase, int h) {
    const unsigned int* p = reinterpret_cast<const unsigned int*>(row);
    u32x8 t;
#pragma unroll
    for (int v = 0; v < 8; ++v) t[v] = p[dwbase + (v < 4 ? v : v + 4) + 4 * h];
    return __builtin_bit_cast(bf16x16, t);
}

__device__ __forceinline__ bf16x16 ldA_P(const unsigned short* row, int h) {
    const unsigned int* p = reinterpret_cast<const unsigned int*>(row);
    u32x8 t;
#pragma unroll
    for (int v = 0; v < 8; ++v) t[v] = p[(v < 4 ? v : v + 4) + 4 * h];
    return __builtin_bit_cast(bf16x16, t);
}

// 16x16 score tile: D = Ks_strip(16 x 128) * Qs_tile^T(128 x 16), f32 accumulate
__device__ __forceinline__ f32x8 score_tile(const unsigned short* Ksh,
                                            const unsigned short* Qsh,
                                            int mbase, int jrel, int n16, int h) {
    f32x8 acc = {};
    const unsigned short* arow = Ksh + (mbase + n16) * KS_STRIDE;
    const unsigned short* brow = Qsh + (jrel + n16) * QS_STRIDE;
#pragma unroll
    for (int kc = 0; kc < 4; ++kc) {
        bf16x16 a = ldA(arow, kc, h);
        bf16x16 b = ldB(brow, kc, h);
        acc = __builtin_amdgcn_wmma_f32_16x16x32_bf16(false, a, false, b,
                                                      (short)0, acc, false, false);
    }
    return acc;
}

__global__ __launch_bounds__(NTHREADS)
void dpa_fwd_kernel(const float* __restrict__ Q, const float* __restrict__ K,
                    const float* __restrict__ V, float* __restrict__ outO,
                    float* __restrict__ outW) {
    __shared__ unsigned short Ks[BM * KS_STRIDE];     // K rows (pre-scaled), resident
    __shared__ unsigned short Qs[BM * QS_STRIDE];     // Q tile (B-matrix of QK)
    __shared__ unsigned short Vt[DHEAD * VT_STRIDE];  // V tile transposed (B-matrix of PV)
    __shared__ unsigned short Ps[4][16 * PS_STRIDE];  // per-wave P C->A relayout scratch

    const int tid  = threadIdx.x;
    const int w    = tid >> 5;
    const int lane = tid & 31;
    const int n16  = lane & 15;
    const int h    = lane >> 4;

    const int nb = S_LEN / BM;                        // 32 row-blocks per batch
    const int b  = blockIdx.x / nb;
    const int i0 = (nb - 1 - (blockIdx.x % nb)) * BM; // reverse order: lightest blocks first
    const int mbase = 16 * w;
    const float scale = 0.08838834764831845f;         // 1/sqrt(128)

    // ---- stage K row-block: b128 loads, bf16-pair packed dword stores ------
    for (int t = tid; t < BM * DHEAD / 4; t += NTHREADS) {
        int r = t >> 5, c4 = t & 31;                   // 32 float4 per row
        float4 v4 = *reinterpret_cast<const float4*>(
            K + ((size_t)(b * S_LEN + i0 + r)) * DHEAD + c4 * 4);
        unsigned int* p = reinterpret_cast<unsigned int*>(Ks) + r * (KS_STRIDE / 2) + c4 * 2;
        p[0] = pack2bf(v4.x * scale, v4.y * scale);
        p[1] = pack2bf(v4.z * scale, v4.w * scale);
    }

    // ---- zero-fill weights region j in [0, i0) (exactly 0 in reference) ----
    if (i0 > 0) {
        const int cpr = i0 >> 2;                       // float4 per row
        float4 z = make_float4(0.f, 0.f, 0.f, 0.f);
        for (int t = tid; t < BM * cpr; t += NTHREADS) {
            int r = t / cpr, c4 = t % cpr;
            float4* dst = reinterpret_cast<float4*>(
                outW + ((size_t)(b * S_LEN + i0 + r)) * S_LEN) + c4;
            *dst = z;
        }
    }

    float mrow[8], lrow[8];
#pragma unroll
    for (int r = 0; r < 8; ++r) { mrow[r] = -3.0e38f; lrow[r] = 0.0f; }

    // ======================= Pass 1: row max / row sum =======================
    for (int j0 = i0; j0 < S_LEN; j0 += BM) {
        __syncthreads();
        for (int t = tid; t < BM * DHEAD / 4; t += NTHREADS) {
            int r = t >> 5, c4 = t & 31;
            float4 v4 = *reinterpret_cast<const float4*>(
                Q + ((size_t)(b * S_LEN + j0 + r)) * DHEAD + c4 * 4);
            unsigned int* p = reinterpret_cast<unsigned int*>(Qs) + r * (QS_STRIDE / 2) + c4 * 2;
            p[0] = pack2bf(v4.x, v4.y);
            p[1] = pack2bf(v4.z, v4.w);
        }
        __syncthreads();
#pragma unroll
        for (int jj = 0; jj < 4; ++jj) {
            const int jt = j0 + 16 * jj;
            f32x8 s = score_tile(Ks, Qs, mbase, 16 * jj, n16, h);
#pragma unroll
            for (int r = 0; r < 8; ++r) {
                int row_g = i0 + mbase + r + 8 * h;
                int col_g = jt + n16;
                float sv = (col_g < row_g) ? -1.0e7f : s[r];
                float tm = sv;                          // row max across 16 lanes
                tm = fmaxf(tm, __shfl_xor(tm, 1, 32));
                tm = fmaxf(tm, __shfl_xor(tm, 2, 32));
                tm = fmaxf(tm, __shfl_xor(tm, 4, 32));
                tm = fmaxf(tm, __shfl_xor(tm, 8, 32));
                float nm = fmaxf(mrow[r], tm);
                float p  = __expf(sv - nm);
                float rs = p;                           // row sum
                rs += __shfl_xor(rs, 1, 32);
                rs += __shfl_xor(rs, 2, 32);
                rs += __shfl_xor(rs, 4, 32);
                rs += __shfl_xor(rs, 8, 32);
                lrow[r] = lrow[r] * __expf(mrow[r] - nm) + rs;
                mrow[r] = nm;
            }
        }
    }

    float linv[8];
#pragma unroll
    for (int r = 0; r < 8; ++r) linv[r] = 1.0f / lrow[r];

    f32x8 o[8];
#pragma unroll
    for (int dt = 0; dt < 8; ++dt) o[dt] = f32x8{};

    // ============ Pass 2: recompute, emit weights, accumulate P*V ============
    for (int j0 = i0; j0 < S_LEN; j0 += BM) {
        __syncthreads();
        for (int t = tid; t < BM * DHEAD / 4; t += NTHREADS) {
            int r = t >> 5, c4 = t & 31;
            float4 v4 = *reinterpret_cast<const float4*>(
                Q + ((size_t)(b * S_LEN + j0 + r)) * DHEAD + c4 * 4);
            unsigned int* p = reinterpret_cast<unsigned int*>(Qs) + r * (QS_STRIDE / 2) + c4 * 2;
            p[0] = pack2bf(v4.x, v4.y);
            p[1] = pack2bf(v4.z, v4.w);
        }
        for (int t = tid; t < BM * DHEAD / 4; t += NTHREADS) {
            int j = t >> 5, c4 = t & 31;                // transpose into Vt[d][j]
            float4 v4 = *reinterpret_cast<const float4*>(
                V + ((size_t)(b * S_LEN + j0 + j)) * DHEAD + c4 * 4);
            Vt[(c4 * 4 + 0) * VT_STRIDE + j] = f2bf(v4.x);
            Vt[(c4 * 4 + 1) * VT_STRIDE + j] = f2bf(v4.y);
            Vt[(c4 * 4 + 2) * VT_STRIDE + j] = f2bf(v4.z);
            Vt[(c4 * 4 + 3) * VT_STRIDE + j] = f2bf(v4.w);
        }
        __syncthreads();
#pragma unroll
        for (int jc = 0; jc < 2; ++jc) {                // 32-wide K-chunks for PV
#pragma unroll
            for (int js = 0; js < 2; ++js) {
                const int jrel = jc * 32 + js * 16;
                const int jt   = j0 + jrel;
                f32x8 s = score_tile(Ks, Qs, mbase, jrel, n16, h);
#pragma unroll
                for (int r = 0; r < 8; ++r) {
                    int row_g = i0 + mbase + r + 8 * h;
                    int col_g = jt + n16;
                    float sv = (col_g < row_g) ? -1.0e7f : s[r];
                    float p  = __expf(sv - mrow[r]) * linv[r];
                    outW[((size_t)(b * S_LEN + row_g)) * S_LEN + col_g] = p;
                    Ps[w][(r + 8 * h) * PS_STRIDE + js * 16 + n16] = f2bf(p);
                }
            }
            // wave-private LDS RAW: LDS is in-order per wave; block compiler reordering
            asm volatile("s_wait_dscnt 0" ::: "memory");
            bf16x16 aP = ldA_P(&Ps[w][n16 * PS_STRIDE], h);
#pragma unroll
            for (int dt = 0; dt < 8; ++dt) {
                bf16x16 bV = ldB_at(&Vt[(dt * 16 + n16) * VT_STRIDE], jc * 16, h);
                o[dt] = __builtin_amdgcn_wmma_f32_16x16x32_bf16(false, aP, false, bV,
                                                                (short)0, o[dt],
                                                                false, false);
            }
        }
    }

    // ---- write outputs ------------------------------------------------------
#pragma unroll
    for (int dt = 0; dt < 8; ++dt) {
#pragma unroll
        for (int r = 0; r < 8; ++r) {
            int row_g = i0 + mbase + r + 8 * h;
            outO[((size_t)(b * S_LEN + row_g)) * DHEAD + dt * 16 + n16] = o[dt][r];
        }
    }
}

extern "C" void kernel_launch(void* const* d_in, const int* in_sizes, int n_in,
                              void* d_out, int out_size, void* d_ws, size_t ws_size,
                              hipStream_t stream) {
    (void)in_sizes; (void)n_in; (void)d_ws; (void)ws_size; (void)out_size;
    const float* Q = (const float*)d_in[0];
    const float* K = (const float*)d_in[1];
    const float* V = (const float*)d_in[2];
    float* outO = (float*)d_out;                                   // [B,S,D]
    float* outW = outO + (size_t)NBATCH * S_LEN * DHEAD;           // [B,S,S]
    dim3 grid(NBATCH * (S_LEN / BM));                              // 512 workgroups
    dim3 block(NTHREADS);                                          // 4 x wave32
    dpa_fwd_kernel<<<grid, block, 0, stream>>>(Q, K, V, outO, outW);
}
